// CheckEmotion_1958505087434
// MI455X (gfx1250) — matrix operand order repaired and checked by reading, not compiled, not run on hardware
//
#include <hip/hip_runtime.h>

typedef __bf16 bf16;
typedef __attribute__((ext_vector_type(16))) __bf16 v16bf;
typedef __attribute__((ext_vector_type(8)))  float  v8f;

#define T_STEPS 1024
#define B_TOT   512
#define E_DIM   64
#define H_DIM   128
#define G4H     512
#define BT_TILES 32   // 512 batch rows / 16 rows per tile

// ---------------- workspace layout (bytes) ----------------
// xeA    : 32 * 1024 * 2 frags * 512 halves * 2B = 67,108,864
// wB     : 32 nb * 6 kf * 512 halves * 2B       =    196,608
// bias   : 512 * 4                               =      2,048
// hFinal : 512 * 128 * 4                         =    262,144
#define WS_XEA_OFF   0
#define WS_WB_OFF    67108864ull
#define WS_BIAS_OFF  (WS_WB_OFF + 196608ull)
#define WS_HFIN_OFF  (WS_BIAS_OFF + 2048ull)

// ---------------------------------------------------------------------------
// Pack [W_ih | W_hh] (gates = x@W_ihT + h@W_hhT, W stored [N=4H, K]) into
// bf16 WMMA B-fragments: per (nb, kf) fragment of 32x16 (KxN):
//   lane L: N = nb*16 + (L&15), K = (e<8 ? (L>>4)*8+e : 16+(L>>4)*8+(e-8)),  e = 0..15
// kf 0..1 -> W_ih K-blocks (E=64), kf 2..5 -> W_hh K-blocks (H=128).
// Also bias = b_ih + b_hh.
// ---------------------------------------------------------------------------
__global__ void pack_weights(const float* __restrict__ w_ih, const float* __restrict__ w_hh,
                             const float* __restrict__ b_ih, const float* __restrict__ b_hh,
                             bf16* __restrict__ wB, float* __restrict__ bias) {
    int tid = blockIdx.x * blockDim.x + threadIdx.x;   // 32*6*512 = 98304 threads
    if (tid < G4H) bias[tid] = b_ih[tid] + b_hh[tid];
    if (tid >= 32 * 6 * 512) return;
    int frag = tid >> 9;
    int p    = tid & 511;
    int lane = p >> 4, e = p & 15;
    int nb = frag / 6, kf = frag % 6;
    int N = nb * 16 + (lane & 15);
    int k = (e < 8) ? (((lane >> 4) << 3) + e) : (16 + ((lane >> 4) << 3) + (e - 8));
    float v = (kf < 2) ? w_ih[N * E_DIM + kf * 32 + k]
                       : w_hh[N * H_DIM + (kf - 2) * 32 + k];
    wB[tid] = (bf16)v;
}

// ---------------------------------------------------------------------------
// Embedding gather + pack into bf16 WMMA A-fragments (16x32, per ISA layout):
//   lane L = hi*16 + M ; element e:  K = kf*32 + (e<8 ? hi*8+e : 16+hi*8+(e-8))
// One output element per thread. xeA[(bt*1024 + t)*1024 + kf*512 + lane*16 + e]
// ---------------------------------------------------------------------------
__global__ void pack_xe(const long long* __restrict__ x, const float* __restrict__ emb,
                        bf16* __restrict__ xeA) {
    long long tid = (long long)blockIdx.x * blockDim.x + threadIdx.x; // 33,554,432
    int e    = (int)(tid & 15);
    int lane = (int)((tid >> 4) & 31);
    int kf   = (int)((tid >> 9) & 1);
    int t    = (int)((tid >> 10) & 1023);
    int bt   = (int)(tid >> 20);
    int m = lane & 15, hi = lane >> 4;
    int kk  = (e < 8) ? (hi * 8 + e) : (16 + hi * 8 + (e - 8));
    int col = kf * 32 + kk;                 // E index 0..63
    int b   = bt * 16 + m;
    long long tok = x[(long long)b * T_STEPS + t];
    xeA[tid] = (bf16)emb[tok * E_DIM + col];
}

// ---------------------------------------------------------------------------
// Fast device nonlinearities via v_exp_f32 / v_rcp_f32.
// ---------------------------------------------------------------------------
__device__ __forceinline__ float fast_sig(float xv) {
    float ex = __builtin_amdgcn_exp2f(-1.442695041f * xv);
    return __builtin_amdgcn_rcpf(1.0f + ex);
}
__device__ __forceinline__ float fast_tanh(float xv) {
    return 2.0f * fast_sig(2.0f * xv) - 1.0f;
}

// ---------------------------------------------------------------------------
// Main LSTM scan. Grid = 32 blocks (one per 16-row batch tile), 256 thr = 8
// waves. Wave w owns gate N-tiles {w, 8+w, 16+w, 24+w} (i,f,g,o) and keeps
// their 24 B-fragments in VGPRs for all 1024 steps. h shared cross-wave via
// 8KB double-buffered LDS in A-fragment layout; c,h state stays in f32 regs.
// ---------------------------------------------------------------------------
__global__ void __launch_bounds__(256, 1)
lstm_main(const bf16* __restrict__ xeA, const bf16* __restrict__ wBg,
          const float* __restrict__ bias, float* __restrict__ hFinal) {
    __shared__ bf16 hA[2][4 * 512];        // 2 buffers x 4 K-fragments of h

    const int lane = threadIdx.x & 31;
    const int w    = threadIdx.x >> 5;
    const int bt   = blockIdx.x;

    // zero both h buffers (h0 = 0)
    for (int i = threadIdx.x; i < 2 * 2048; i += 256) ((bf16*)hA)[i] = (bf16)0.0f;

    // ---- load loop-invariant weight fragments into registers ----
    v16bf wB[4][6];
    float bs[4];
    const int nbs[4] = { w, 8 + w, 16 + w, 24 + w };
#pragma unroll
    for (int ti = 0; ti < 4; ++ti) {
        bs[ti] = bias[nbs[ti] * 16 + (lane & 15)];
#pragma unroll
        for (int kf = 0; kf < 6; ++kf)
            wB[ti][kf] = *(const v16bf*)(wBg + ((nbs[ti] * 6 + kf) * 512 + lane * 16));
    }

    // ---- per-lane h scatter address (C-layout -> A-fragment layout) ----
    const int m0    = 8 * (lane >> 4);          // row base for acc element r
    const int cglob = w * 16 + (lane & 15);     // h column this lane produces
    const int kb    = cglob >> 5;
    const int o     = cglob & 31;
    const int e_h   = (o < 16) ? (o & 7) : (8 + (o & 7));
    const int hi_h  = (o & 15) >> 3;
    const int hbase = kb * 512 + (hi_h * 16 + m0) * 16 + e_h;   // + r*16 per r

    // ---- f32 recurrent state in registers ----
    v8f cc, hv;
#pragma unroll
    for (int r = 0; r < 8; ++r) { cc[r] = 0.0f; hv[r] = 0.0f; }

    const bf16* xp = xeA + (long long)bt * T_STEPS * 1024;
    v16bf a0 = *(const v16bf*)(xp + lane * 16);
    v16bf a1 = *(const v16bf*)(xp + 512 + lane * 16);

    __syncthreads();    // hA zero-init visible
    int cur = 0;

    for (int t = 0; t < T_STEPS; ++t) {
        // prefetch next step's xe fragments while this step computes
        v16bf n0 = a0, n1 = a1;
        if (t + 1 < T_STEPS) {
            const bf16* np = xp + (long long)(t + 1) * 1024;
            n0 = *(const v16bf*)(np + lane * 16);
            n1 = *(const v16bf*)(np + 512 + lane * 16);
        }

        // gather h A-fragments from LDS (zero at t=0)
        v16bf h0 = *(const v16bf*)(&hA[cur][0 * 512 + lane * 16]);
        v16bf h1 = *(const v16bf*)(&hA[cur][1 * 512 + lane * 16]);
        v16bf h2 = *(const v16bf*)(&hA[cur][2 * 512 + lane * 16]);
        v16bf h3 = *(const v16bf*)(&hA[cur][3 * 512 + lane * 16]);

        // gates[16x16] per tile: bias + xe(K=64) + h(K=128), bf16 WMMA f32 acc
        v8f acc[4];
#pragma unroll
        for (int ti = 0; ti < 4; ++ti) {
            v8f a;
#pragma unroll
            for (int r = 0; r < 8; ++r) a[r] = bs[ti];
            a = __builtin_amdgcn_wmma_f32_16x16x32_bf16(false, a0, false, wB[ti][0], (short)0, a, false, false);
            a = __builtin_amdgcn_wmma_f32_16x16x32_bf16(false, a1, false, wB[ti][1], (short)0, a, false, false);
            a = __builtin_amdgcn_wmma_f32_16x16x32_bf16(false, h0, false, wB[ti][2], (short)0, a, false, false);
            a = __builtin_amdgcn_wmma_f32_16x16x32_bf16(false, h1, false, wB[ti][3], (short)0, a, false, false);
            a = __builtin_amdgcn_wmma_f32_16x16x32_bf16(false, h2, false, wB[ti][4], (short)0, a, false, false);
            a = __builtin_amdgcn_wmma_f32_16x16x32_bf16(false, h3, false, wB[ti][5], (short)0, a, false, false);
            acc[ti] = a;
        }

        const int nxt = cur ^ 1;
#pragma unroll
        for (int r = 0; r < 8; ++r) {
            float iv = fast_sig(acc[0][r]);
            float fv = fast_sig(acc[1][r]);
            float gv = fast_tanh(acc[2][r]);
            float ov = fast_sig(acc[3][r]);
            float c  = fv * cc[r] + iv * gv;
            cc[r] = c;
            float h = ov * fast_tanh(c);
            hv[r] = h;
            hA[nxt][hbase + r * 16] = (bf16)h;   // scatter into A-fragment layout
        }
        __syncthreads();     // separates this step's reads/writes from next step
        a0 = n0; a1 = n1;
        cur = nxt;
    }

    // final h in fp32 for the classifier head
#pragma unroll
    for (int r = 0; r < 8; ++r) {
        int brow = bt * 16 + m0 + r;
        hFinal[brow * H_DIM + cglob] = hv[r];
    }
}

// ---------------------------------------------------------------------------
// out[b, c] = hFinal[b,:] . w_lin[c,:] + b_lin[c]   (512 x 28, tiny)
// ---------------------------------------------------------------------------
__global__ void final_linear(const float* __restrict__ hFinal, const float* __restrict__ w_lin,
                             const float* __restrict__ b_lin, float* __restrict__ out) {
    int tid = blockIdx.x * blockDim.x + threadIdx.x;
    if (tid >= B_TOT * 28) return;
    int b = tid / 28, c = tid % 28;
    const float* hr = hFinal + b * H_DIM;
    const float* wr = w_lin + c * H_DIM;
    float s = b_lin[c];
#pragma unroll 8
    for (int k = 0; k < H_DIM; ++k) s += hr[k] * wr[k];
    out[tid] = s;
}

extern "C" void kernel_launch(void* const* d_in, const int* in_sizes, int n_in,
                              void* d_out, int out_size, void* d_ws, size_t ws_size,
                              hipStream_t stream) {
    const long long* x     = (const long long*)d_in[0];
    const float*     emb   = (const float*)d_in[1];
    const float*     w_ih  = (const float*)d_in[2];
    const float*     w_hh  = (const float*)d_in[3];
    const float*     b_ih  = (const float*)d_in[4];
    const float*     b_hh  = (const float*)d_in[5];
    const float*     w_lin = (const float*)d_in[6];
    const float*     b_lin = (const float*)d_in[7];
    float* out = (float*)d_out;

    char*  ws     = (char*)d_ws;
    bf16*  xeA    = (bf16*)(ws + WS_XEA_OFF);
    bf16*  wB     = (bf16*)(ws + WS_WB_OFF);
    float* bias   = (float*)(ws + WS_BIAS_OFF);
    float* hFinal = (float*)(ws + WS_HFIN_OFF);

    // 1) pack weights + bias (98304 threads)
    pack_weights<<<384, 256, 0, stream>>>(w_ih, w_hh, b_ih, b_hh, wB, bias);
    // 2) embedding gather + bf16 A-fragment pack (33.5M threads, ~134MB traffic)
    pack_xe<<<131072, 256, 0, stream>>>(x, emb, xeA);
    // 3) fused LSTM scan: 32 independent batch tiles x 1024 steps
    lstm_main<<<BT_TILES, 256, 0, stream>>>(xeA, wB, bias, hFinal);
    // 4) classifier head
    final_linear<<<(B_TOT * 28 + 255) / 256, 256, 0, stream>>>(hFinal, w_lin, b_lin, out);
}
